// EGCN_O_70042326663890
// MI455X (gfx1250) — matrix operand order repaired and checked by reading, not compiled, not run on hardware
//
#include <hip/hip_runtime.h>

typedef __attribute__((ext_vector_type(16))) __bf16 bf16x16;
typedef __attribute__((ext_vector_type(8)))  __bf16 bf16x8;
typedef __attribute__((ext_vector_type(8)))  float  f32x8;
typedef __attribute__((ext_vector_type(4)))  float  f32x4;
typedef __attribute__((ext_vector_type(4)))  int    i32x4;

#define BM 128
#define BN 64
#define BK 64
#define LPITCH 72   // 64 + 8 bf16 pad; row stride 144 B (16B-aligned)

#if defined(__has_builtin)
#if __has_builtin(__builtin_amdgcn_global_load_async_to_lds_b128) && \
    __has_builtin(__builtin_amdgcn_s_wait_asynccnt)
#define USE_ASYNC 1
#endif
#endif

__device__ __forceinline__ void copy16(const __bf16* g, __bf16* l)
{
#ifdef USE_ASYNC
    __builtin_amdgcn_global_load_async_to_lds_b128(
        (__attribute__((address_space(1))) i32x4*)(__bf16*)g,
        (__attribute__((address_space(3))) i32x4*)l, 0, 0);
#else
    *(bf16x8*)l = *(const bf16x8*)g;
#endif
}

// C = act(A @ B). A: [M,K] bf16 row-major. Bt: [N,K] bf16 row-major (= B^T).
// mode 0: C f32 row-major; mode 1: C bf16 row-major + relu; mode 2: C^T bf16
// ([N,M] row-major, 16B vector stores). M%128==0, N%64==0, K%64==0.
__global__ __launch_bounds__(256)
void wmma_gemm_bf16(const __bf16* __restrict__ A, const __bf16* __restrict__ Bt,
                    void* __restrict__ Cv, int M, int K, int N, int mode)
{
    __shared__ __bf16 As[2][BM * LPITCH];
    __shared__ __bf16 Bs[2][BN * LPITCH];

    const int tid   = threadIdx.x;
    const int lane  = tid & 31;
    const int wave  = tid >> 5;
    const int waveM = wave >> 1;     // 0..3
    const int waveN = wave & 1;      // 0..1
    const int hl    = lane >> 4;
    const int l16   = lane & 15;

    const int bm = blockIdx.x * BM;
    const int bn = blockIdx.y * BN;

    // A tile 128x64 bf16: 2 threads/row, 4x16B chunks each.
    const int arow = tid >> 1;
    const int acol = (tid & 1) << 5;
    // B^T tile 64x64 bf16: 4 threads/row, 2x16B chunks each.
    const int brow = tid >> 2;
    const int bcol = (tid & 3) << 4;

    f32x8 acc[2][2];
#pragma unroll
    for (int i = 0; i < 2; ++i)
#pragma unroll
        for (int j = 0; j < 2; ++j)
#pragma unroll
            for (int e = 0; e < 8; ++e) acc[i][j][e] = 0.0f;

    auto issue = [&](int buf, int k0) {
        const __bf16* ga = A + (size_t)(bm + arow) * K + k0 + acol;
        __bf16* la = &As[buf][arow * LPITCH + acol];
#pragma unroll
        for (int i = 0; i < 4; ++i) copy16(ga + i * 8, la + i * 8);
        const __bf16* gb = Bt + (size_t)(bn + brow) * K + k0 + bcol;
        __bf16* lb = &Bs[buf][brow * LPITCH + bcol];
#pragma unroll
        for (int i = 0; i < 2; ++i) copy16(gb + i * 8, lb + i * 8);
    };

    issue(0, 0);
#ifdef USE_ASYNC
    __builtin_amdgcn_s_wait_asynccnt(0);
#endif
    __syncthreads();

    const int nk = K / BK;
    int cur = 0;
    for (int kt = 0; kt < nk; ++kt) {
        const bool more = (kt + 1) < nk;
        if (more) issue(cur ^ 1, (kt + 1) * BK);   // async into other buffer

#pragma unroll
        for (int ks = 0; ks < 2; ++ks) {           // two 16x16x32 K-phases
            const int kb = ks * 32;
            bf16x16 af[2], bfv[2];
#pragma unroll
            for (int i = 0; i < 2; ++i) {
                const int m = waveM * 32 + i * 16 + l16;
                bf16x8 lo = *(const bf16x8*)&As[cur][m * LPITCH + kb + hl * 8];
                bf16x8 hi = *(const bf16x8*)&As[cur][m * LPITCH + kb + 16 + hl * 8];
#pragma unroll
                for (int e = 0; e < 8; ++e) { af[i][e] = lo[e]; af[i][8 + e] = hi[e]; }
            }
#pragma unroll
            for (int j = 0; j < 2; ++j) {
                const int n = waveN * 32 + j * 16 + l16;
                bf16x8 lo = *(const bf16x8*)&Bs[cur][n * LPITCH + kb + hl * 8];
                bf16x8 hi = *(const bf16x8*)&Bs[cur][n * LPITCH + kb + 16 + hl * 8];
#pragma unroll
                for (int e = 0; e < 8; ++e) { bfv[j][e] = lo[e]; bfv[j][8 + e] = hi[e]; }
            }
#pragma unroll
            for (int i = 0; i < 2; ++i)
#pragma unroll
                for (int j = 0; j < 2; ++j)
                    acc[i][j] = __builtin_amdgcn_wmma_f32_16x16x32_bf16(
                        false, af[i], false, bfv[j], (short)0, acc[i][j], false, false);
        }

#ifdef USE_ASYNC
        if (more) __builtin_amdgcn_s_wait_asynccnt(0);
#endif
        __syncthreads();
        cur ^= 1;
    }

    // Epilogue. C/D layout: VGPR r -> row (r + hl*8), col = l16.
#pragma unroll
    for (int i = 0; i < 2; ++i) {
#pragma unroll
        for (int j = 0; j < 2; ++j) {
            const int gm = bm + waveM * 32 + i * 16 + hl * 8;
            const int gn = bn + waveN * 32 + j * 16 + l16;
            if (mode == 0) {
                float* C = (float*)Cv;
                float* cp = C + (size_t)gm * N + gn;
#pragma unroll
                for (int r = 0; r < 8; ++r) cp[(size_t)r * N] = acc[i][j][r];
            } else if (mode == 1) {
                __bf16* C = (__bf16*)Cv;
#pragma unroll
                for (int r = 0; r < 8; ++r)
                    C[(size_t)(gm + r) * N + gn] = (__bf16)fmaxf(acc[i][j][r], 0.0f);
            } else {
                __bf16* CT = (__bf16*)Cv;      // [N, M] row-major; 8 rows contiguous
                bf16x8 p;
#pragma unroll
                for (int r = 0; r < 8; ++r) p[r] = (__bf16)acc[i][j][r];
                *(bf16x8*)&CT[(size_t)gn * M + gm] = p;   // global_store_b128
            }
        }
    }
}

// Elementwise f32 -> bf16 (row-major), n % 8 == 0.
__global__ void cvt_bf16(const float* __restrict__ in, __bf16* __restrict__ out, size_t n)
{
    const size_t i = ((size_t)blockIdx.x * blockDim.x + threadIdx.x) * 8;
    if (i >= n) return;
    f32x4 a = *(const f32x4*)(in + i);
    f32x4 b = *(const f32x4*)(in + i + 4);
    bf16x8 o;
#pragma unroll
    for (int e = 0; e < 4; ++e) { o[e] = (__bf16)a[e]; o[e + 4] = (__bf16)b[e]; }
    *(bf16x8*)(out + i) = o;
}

// out[c*R + r] = (bf16) in[r*C + c]   (f32 [R,C] -> bf16 transpose [C,R])
__global__ void cvt_bf16_tr(const float* __restrict__ in, __bf16* __restrict__ out,
                            int R, int C)
{
    const int id = blockIdx.x * blockDim.x + threadIdx.x;
    if (id >= R * C) return;
    const int r = id / C, c = id % C;
    out[(size_t)c * R + r] = (__bf16)in[id];
}

// GRU cell phase 1: U = sigmoid((W0+U0)@Q + B0), R = sigmoid((W1+U1)@Q + B1)
// (z == Q), S = W2@Q.
__global__ void cell_phase1(const float* __restrict__ Wg, const float* __restrict__ Ug,
                            const float* __restrict__ Bg, const float* __restrict__ Q,
                            float* __restrict__ U, float* __restrict__ R,
                            float* __restrict__ S, int F, int C)
{
    const int id = blockIdx.x * blockDim.x + threadIdx.x;
    if (id >= F * C) return;
    const int c = id % C;
    const int r = id / C;
    const float* W0 = Wg;
    const float* W1 = Wg + (size_t)F * F;
    const float* W2 = Wg + (size_t)2 * F * F;
    const float* U0 = Ug;
    const float* U1 = Ug + (size_t)F * F;
    float s0 = 0.f, s1 = 0.f, s2 = 0.f;
    for (int k = 0; k < F; ++k) {
        const float q = Q[(size_t)k * C + c];
        s0 += (W0[(size_t)r * F + k] + U0[(size_t)r * F + k]) * q;
        s1 += (W1[(size_t)r * F + k] + U1[(size_t)r * F + k]) * q;
        s2 += W2[(size_t)r * F + k] * q;
    }
    U[id] = 1.0f / (1.0f + __expf(-(s0 + Bg[id])));
    R[id] = 1.0f / (1.0f + __expf(-(s1 + Bg[(size_t)F * C + id])));
    S[id] = s2;
}

// Phase 2: h~ = tanh(S + U2@(R*Q) + B2); Qn = (1-U)*Q + U*h~
__global__ void cell_phase2(const float* __restrict__ Ug, const float* __restrict__ Bg,
                            const float* __restrict__ Q, const float* __restrict__ U,
                            const float* __restrict__ R, const float* __restrict__ S,
                            float* __restrict__ Qn, int F, int C)
{
    const int id = blockIdx.x * blockDim.x + threadIdx.x;
    if (id >= F * C) return;
    const int c = id % C;
    const int r = id / C;
    const float* U2 = Ug + (size_t)2 * F * F;
    float t = 0.f;
    for (int k = 0; k < F; ++k)
        t += U2[(size_t)r * F + k] * (R[(size_t)k * C + c] * Q[(size_t)k * C + c]);
    const float hc = tanhf(S[id] + t + Bg[(size_t)2 * F * C + id]);
    const float u  = U[id];
    Qn[id] = (1.0f - u) * Q[id] + u * hc;
}

extern "C" void kernel_launch(void* const* d_in, const int* in_sizes, int n_in,
                              void* d_out, int out_size, void* d_ws, size_t ws_size,
                              hipStream_t stream)
{
    (void)in_sizes; (void)n_in; (void)out_size; (void)ws_size;
    constexpr int T = 8, N = 4096, F0 = 256, F1 = 128, F2 = 64;

    const float* adj   = (const float*)d_in[0];
    const float* feats = (const float*)d_in[1];
    const float* Wg1   = (const float*)d_in[2];
    const float* Ug1   = (const float*)d_in[3];
    const float* Bg1   = (const float*)d_in[4];
    const float* Q1    = (const float*)d_in[5];
    const float* Wg2   = (const float*)d_in[6];
    const float* Ug2   = (const float*)d_in[7];
    const float* Bg2   = (const float*)d_in[8];
    const float* Q2    = (const float*)d_in[9];

    char* wp = (char*)d_ws;
    auto alloc = [&](size_t bytes) -> void* {
        void* r = (void*)wp;
        wp += (bytes + 255) & ~(size_t)255;
        return r;
    };
    float*  Qa1   = (float*)alloc(sizeof(float) * F0 * F1);
    float*  Qb1   = (float*)alloc(sizeof(float) * F0 * F1);
    float*  Ub1   = (float*)alloc(sizeof(float) * F0 * F1);
    float*  Rb1   = (float*)alloc(sizeof(float) * F0 * F1);
    float*  Sb1   = (float*)alloc(sizeof(float) * F0 * F1);
    float*  Qa2   = (float*)alloc(sizeof(float) * F1 * F2);
    float*  Qb2   = (float*)alloc(sizeof(float) * F1 * F2);
    float*  Ub2   = (float*)alloc(sizeof(float) * F1 * F2);
    float*  Rb2   = (float*)alloc(sizeof(float) * F1 * F2);
    float*  Sb2   = (float*)alloc(sizeof(float) * F1 * F2);
    __bf16* adjbf = (__bf16*)alloc((size_t)N * N * 2);
    __bf16* fbf   = (__bf16*)alloc((size_t)N * F0 * 2);
    __bf16* Q1T   = (__bf16*)alloc((size_t)F1 * F0 * 2);
    __bf16* Q2T   = (__bf16*)alloc((size_t)F2 * F1 * 2);
    __bf16* X1T   = (__bf16*)alloc((size_t)F1 * N * 2);
    __bf16* H1b   = (__bf16*)alloc((size_t)N * F1 * 2);
    __bf16* X2T   = (__bf16*)alloc((size_t)F2 * N * 2);

    (void)hipMemcpyAsync(Qa1, Q1, sizeof(float) * F0 * F1, hipMemcpyDeviceToDevice, stream);
    (void)hipMemcpyAsync(Qa2, Q2, sizeof(float) * F1 * F2, hipMemcpyDeviceToDevice, stream);

    // Only h2[-1] is needed: evolve Q1/Q2 8x (independent of adj/feats), then
    // do the t=7 embedding pipeline once.
    float *c1 = Qa1, *n1 = Qb1, *c2 = Qa2, *n2 = Qb2;
    for (int t = 0; t < T; ++t) {
        cell_phase1<<<(F0 * F1) / 256, 256, 0, stream>>>(Wg1, Ug1, Bg1, c1, Ub1, Rb1, Sb1, F0, F1);
        cell_phase2<<<(F0 * F1) / 256, 256, 0, stream>>>(Ug1, Bg1, c1, Ub1, Rb1, Sb1, n1, F0, F1);
        float* tmp = c1; c1 = n1; n1 = tmp;
        cell_phase1<<<(F1 * F2) / 256, 256, 0, stream>>>(Wg2, Ug2, Bg2, c2, Ub2, Rb2, Sb2, F1, F2);
        cell_phase2<<<(F1 * F2) / 256, 256, 0, stream>>>(Ug2, Bg2, c2, Ub2, Rb2, Sb2, n2, F1, F2);
        tmp = c2; c2 = n2; n2 = tmp;
    }

    const float* adj7 = adj   + (size_t)(T - 1) * N * N;
    const float* f7   = feats + (size_t)(T - 1) * N * F0;

    // One-time bf16 conversions (bandwidth-cheap vs compute-bound GEMMs).
    cvt_bf16<<<((size_t)N * N / 8 + 255) / 256, 256, 0, stream>>>(adj7, adjbf, (size_t)N * N);
    cvt_bf16<<<((size_t)N * F0 / 8 + 255) / 256, 256, 0, stream>>>(f7, fbf, (size_t)N * F0);
    cvt_bf16_tr<<<(F0 * F1) / 256, 256, 0, stream>>>(c1, Q1T, F0, F1);
    cvt_bf16_tr<<<(F1 * F2) / 256, 256, 0, stream>>>(c2, Q2T, F1, F2);

    // X1^T = (feats7 @ Q1)^T        [4096,256]x[256,128] -> bf16 [128,4096]
    wmma_gemm_bf16<<<dim3(N / BM, F1 / BN), 256, 0, stream>>>(fbf, Q1T, X1T, N, F0, F1, 2);
    // H1 = relu(adj7 @ X1)          [4096,4096]x[4096,128] -> bf16 [4096,128]
    wmma_gemm_bf16<<<dim3(N / BM, F1 / BN), 256, 0, stream>>>(adjbf, X1T, H1b, N, N, F1, 1);
    // X2^T = (H1 @ Q2)^T            [4096,128]x[128,64] -> bf16 [64,4096]
    wmma_gemm_bf16<<<dim3(N / BM, F2 / BN), 256, 0, stream>>>(H1b, Q2T, X2T, N, F1, F2, 2);
    // out = adj7 @ X2               [4096,4096]x[4096,64] -> f32 d_out
    wmma_gemm_bf16<<<dim3(N / BM, F2 / BN), 256, 0, stream>>>(adjbf, X2T, d_out, N, N, F2, 0);
}
